// TransformerModel_84258668413617
// MI455X (gfx1250) — matrix operand order repaired and checked by reading, not compiled, not run on hardware
//
#include <hip/hip_runtime.h>
#include <hip/hip_bf16.h>
#include <cstdint>

typedef __attribute__((ext_vector_type(16))) __bf16 v16bf;
typedef __attribute__((ext_vector_type(8)))  __bf16 v8bf;
typedef __attribute__((ext_vector_type(8)))  float  v8f;

#define D_MODEL 128
#define N_LAYERS 4
#define D_FF 512
#define D_INNER 256
#define D_STATE 16
#define DT_RANK 8
#define BB 8
#define LL 4096
#define M_ROWS (BB * LL)   // 32768

// ---------------------------------------------------------------------------
// bf16 x bf16^T -> fp32 acc GEMM.  A[M,K] row-major bf16, Bt[N,K] row-major
// bf16 (B pre-transposed).  One wave computes a 32x64 C tile: 8
// v_wmma_f32_16x16x32_bf16 per 32-wide K slab (B fragments reused across the
// two 16-row sub-tiles).  Staging: double-buffered GLOBAL_LOAD_ASYNC_TO_LDS_
// B128 (pure bf16 copy), pipelined on ASYNCcnt.  Requires K % 32 == 0
// (host guarantees; W_dt^T is zero-padded 8->32 in K).
// MODE: 0 none, 1 +bias, 2 relu(+bias), 3 softplus(+bias).  TC: float/__bf16.
// ---------------------------------------------------------------------------
#define LDS_K 40                         // 32 K-halves + 8 pad (bank spread)
#define A_HALVES (32 * LDS_K)            // 1280
#define B_HALVES (64 * LDS_K)            // 2560
#define BUF_HALVES (A_HALVES + B_HALVES) // 3840
#define WAVE_HALVES (2 * BUF_HALVES)     // double buffered -> 7680

template <int MODE, typename TC>
__global__ __launch_bounds__(128) void gemm_bf16_wmma(
    const __bf16* __restrict__ A, const __bf16* __restrict__ Bt,
    TC* __restrict__ C, const float* __restrict__ bias,
    int M, int N, int K, int lda, int ldc)
{
    __shared__ __bf16 smem[4 * WAVE_HALVES];   // 61.4 KB, 4 waves/block
    const int lane = threadIdx.x;              // 0..31
    const int wid  = threadIdx.y;              // 0..3

    const int tiles_n = (N + 63) >> 6;
    const int tiles_m = (M + 31) >> 5;
    const int total   = tiles_m * tiles_n;
    int tile = blockIdx.x * 4 + wid;
    const bool active = (tile < total);
    if (!active) tile = 0;
    const int tm = tile / tiles_n;
    const int tn = tile - tm * tiles_n;
    const int m0 = tm << 5;
    const int n0 = tn << 6;

    __bf16* buf0 = &smem[wid * WAVE_HALVES];

    v8f acc[2][4] = {};
    const int fr = lane & 15;              // fragment row / col
    const int ka = (lane >> 4) << 3;       // A frag K base: 0 or 8 halves
    const int kb = (lane >> 4) << 3;       // B frag K base: 0 or 8 halves

    // fragment gather + 8 WMMAs for one staged 32-K slab
    auto mma = [&](const __bf16* As, const __bf16* Bs) {
        v16bf af[2];
        #pragma unroll
        for (int s = 0; s < 2; ++s) {
            const __bf16* ap = As + (s * 16 + fr) * LDS_K + ka;
            v8bf alo = *(const v8bf*)ap;
            v8bf ahi = *(const v8bf*)(ap + 16);
            af[s] = __builtin_shufflevector(alo, ahi,
                0, 1, 2, 3, 4, 5, 6, 7, 8, 9, 10, 11, 12, 13, 14, 15);
        }
        #pragma unroll
        for (int c = 0; c < 4; ++c) {
            const __bf16* bp = Bs + (c * 16 + fr) * LDS_K + kb;
            v8bf blo = *(const v8bf*)bp;
            v8bf bhi = *(const v8bf*)(bp + 16);
            v16bf bv = __builtin_shufflevector(blo, bhi,
                0, 1, 2, 3, 4, 5, 6, 7, 8, 9, 10, 11, 12, 13, 14, 15);
            acc[0][c] = __builtin_amdgcn_wmma_f32_16x16x32_bf16(
                false, af[0], false, bv, (short)0, acc[0][c], false, false);
            acc[1][c] = __builtin_amdgcn_wmma_f32_16x16x32_bf16(
                false, af[1], false, bv, (short)0, acc[1][c], false, false);
        }
    };

    const int T = K >> 5;
    const int lrow = lane >> 2;              // 0..7 (8 rows per async op)
    const unsigned colb = (lane & 3) << 4;   // 0/16/32/48 bytes
    const unsigned base_b = (unsigned)(uintptr_t)buf0;     // LDS byte offset
    const uint64_t Abase = (uint64_t)(uintptr_t)A + (uint64_t)m0 * lda * 2;
    const uint64_t Bbase = (uint64_t)(uintptr_t)Bt + (uint64_t)n0 * K * 2;
    const uint64_t ldab = (uint64_t)lda * 2;
    const uint64_t ldbb = (uint64_t)K * 2;

    auto issue = [&](int t, int sel) {
        const unsigned lb = base_b + (unsigned)sel * (BUF_HALVES * 2);
        const uint64_t kofs = (uint64_t)(t << 5) * 2;
        #pragma unroll
        for (int q = 0; q < 4; ++q) {          // A tile: 32 rows x 64 B
            unsigned lo = lb + (unsigned)(q * 8 + lrow) * (LDS_K * 2) + colb;
            uint64_t ga = Abase + (uint64_t)(q * 8 + lrow) * ldab + kofs + colb;
            asm volatile("global_load_async_to_lds_b128 %0, %1, off"
                         :: "v"(lo), "v"(ga) : "memory");
        }
        #pragma unroll
        for (int q = 0; q < 8; ++q) {          // B tile: 64 rows x 64 B
            unsigned lo = lb + (unsigned)(A_HALVES * 2)
                        + (unsigned)(q * 8 + lrow) * (LDS_K * 2) + colb;
            uint64_t ga = Bbase + (uint64_t)(q * 8 + lrow) * ldbb + kofs + colb;
            asm volatile("global_load_async_to_lds_b128 %0, %1, off"
                         :: "v"(lo), "v"(ga) : "memory");
        }
    };

    issue(0, 0);
    for (int t = 0; t < T; ++t) {
        if (t + 1 < T) {
            issue(t + 1, (t + 1) & 1);
            asm volatile("s_wait_asynccnt 0xc" ::: "memory");  // slab t landed
        } else {
            asm volatile("s_wait_asynccnt 0x0" ::: "memory");
        }
        const __bf16* As = buf0 + (t & 1) * BUF_HALVES;
        mma(As, As + A_HALVES);
    }

    if (!active) return;
    // C layout: VGPR r -> M = r (lanes 0-15) / r+8 (lanes 16-31); N = lane&15
    #pragma unroll
    for (int s = 0; s < 2; ++s) {
        const int rbase = m0 + s * 16 + ((lane >> 4) << 3);
        #pragma unroll
        for (int c = 0; c < 4; ++c) {
            const int col = n0 + c * 16 + fr;
            if (col >= N) continue;
            const float bv = (MODE >= 1) ? bias[col] : 0.f;
            #pragma unroll
            for (int r = 0; r < 8; ++r) {
                float v = acc[s][c][r] + bv;
                if (MODE == 2) v = fmaxf(v, 0.f);
                if (MODE == 3) v = (v > 20.f) ? v : log1pf(__expf(v));
                C[(size_t)(rbase + r) * ldc + col] = (TC)v;
            }
        }
    }
}

// ---------------------------------------------------------------------------
// weight fp32 (K x N) -> bf16 transposed + K-padded (N x Kpad, zeros for k>=K)
__global__ void k_cvtT(const float* __restrict__ src, __bf16* __restrict__ dst,
                       int K, int N, int Kpad)
{
    int i = blockIdx.x * 256 + threadIdx.x;
    if (i >= N * Kpad) return;
    int n = i / Kpad, k = i - n * Kpad;
    dst[i] = (k < K) ? (__bf16)src[(size_t)k * N + n] : (__bf16)0.f;
}

__global__ void k_embed(const int* __restrict__ tok, const float* __restrict__ emb,
                        __bf16* __restrict__ x, int n)
{
    int i = blockIdx.x * 256 + threadIdx.x;
    if (i >= n) return;
    int row = i >> 7, col = i & 127;
    x[i] = (__bf16)emb[(size_t)tok[row] * D_MODEL + col];
}

__global__ __launch_bounds__(256) void k_layernorm(
    const __bf16* __restrict__ x, __bf16* __restrict__ y,
    const float* __restrict__ w, const float* __restrict__ b, int rows)
{
    int row = blockIdx.x * 8 + threadIdx.y;
    if (row >= rows) return;
    int lane = threadIdx.x;
    const __bf16* xr = x + (size_t)row * D_MODEL;
    float v[4], s = 0.f, s2 = 0.f;
    #pragma unroll
    for (int j = 0; j < 4; ++j) {
        v[j] = (float)xr[lane * 4 + j];
        s += v[j]; s2 += v[j] * v[j];
    }
    #pragma unroll
    for (int off = 16; off > 0; off >>= 1) {
        s  += __shfl_xor(s, off, 32);
        s2 += __shfl_xor(s2, off, 32);
    }
    float mu  = s * (1.f / 128.f);
    float var = s2 * (1.f / 128.f) - mu * mu;
    float rs  = rsqrtf(var + 1e-5f);
    #pragma unroll
    for (int j = 0; j < 4; ++j) {
        int c = lane * 4 + j;
        y[(size_t)row * D_MODEL + c] = (__bf16)((v[j] - mu) * rs * w[c] + b[c]);
    }
}

// causal depthwise conv (width 4) + silu; xc = xz[:, :256] (fp32), u bf16
__global__ void k_conv_silu(const float* __restrict__ xz,
                            const float* __restrict__ cw, const float* __restrict__ cb,
                            __bf16* __restrict__ u, int n)
{
    int i = blockIdx.x * 256 + threadIdx.x;
    if (i >= n) return;
    int d  = i & (D_INNER - 1);
    int bl = i >> 8;
    int l  = bl & (LL - 1);
    float acc = cb[d];
    const float* w = cw + d * 4;
    #pragma unroll
    for (int j = 0; j < 4; ++j) {
        int lp = l - 3 + j;
        if (lp >= 0) acc += w[j] * xz[(size_t)(bl - l + lp) * 512 + d];
    }
    u[i] = (__bf16)(acc / (1.f + __expf(-acc)));   // silu
}

// selective scan: 1 block/batch, 1 thread/channel, 16-state regs.
// Barrier-free: every thread broadcast-loads the shared 64 B B/C row.
// uy holds u (bf16) on entry; overwritten with y = (scan + u*D) * silu(z).
__global__ __launch_bounds__(256) void k_scan(
    const float* __restrict__ dt, const __bf16* __restrict__ dbc,
    const float* __restrict__ xz, const float* __restrict__ A_log,
    const float* __restrict__ D_skip, __bf16* __restrict__ uy)
{
    const int b = blockIdx.x;
    const int d = threadIdx.x;
    float Ar[D_STATE], h[D_STATE];
    #pragma unroll
    for (int s = 0; s < D_STATE; ++s) {
        Ar[s] = -__expf(A_log[d * D_STATE + s]);
        h[s]  = 0.f;
    }
    const float Dv = D_skip[d];
    const __bf16* bcp = dbc + (size_t)b * LL * 40 + DT_RANK;   // 16B-aligned
    for (int l = 0; l < LL; ++l) {
        const size_t row = (size_t)b * LL + l;
        v8bf q0 = *(const v8bf*)(bcp + 0);    // B[0:8)
        v8bf q1 = *(const v8bf*)(bcp + 8);    // B[8:16)
        v8bf q2 = *(const v8bf*)(bcp + 16);   // C[0:8)
        v8bf q3 = *(const v8bf*)(bcp + 24);   // C[8:16)
        float Bv[D_STATE], Cv[D_STATE];
        #pragma unroll
        for (int i = 0; i < 8; ++i) {
            Bv[i] = (float)q0[i]; Bv[8 + i] = (float)q1[i];
            Cv[i] = (float)q2[i]; Cv[8 + i] = (float)q3[i];
        }
        const float dtv = dt[row * D_INNER + d];
        const float uv  = (float)uy[row * D_INNER + d];
        const float du  = dtv * uv;
        float y = 0.f;
        #pragma unroll
        for (int s = 0; s < D_STATE; ++s) {
            float dA = __expf(dtv * Ar[s]);
            h[s] = dA * h[s] + du * Bv[s];
            y += h[s] * Cv[s];
        }
        const float z  = xz[row * 512 + D_INNER + d];
        const float sz = z / (1.f + __expf(-z));
        uy[row * D_INNER + d] = (__bf16)((y + uv * Dv) * sz);
        bcp += 40;
    }
}

// ---------------------------------------------------------------------------
extern "C" void kernel_launch(void* const* d_in, const int* in_sizes, int n_in,
                              void* d_out, int out_size, void* d_ws, size_t ws_size,
                              hipStream_t stream)
{
    const int*   tokens = (const int*)  d_in[0];
    const float* emb    = (const float*)d_in[1];
    const float* ln_w   = (const float*)d_in[2];
    const float* ln_b   = (const float*)d_in[3];
    const float* W_in   = (const float*)d_in[4];
    const float* conv_w = (const float*)d_in[5];
    const float* conv_b = (const float*)d_in[6];
    const float* W_xp   = (const float*)d_in[7];
    const float* W_dt   = (const float*)d_in[8];
    const float* b_dt   = (const float*)d_in[9];
    const float* A_log  = (const float*)d_in[10];
    const float* D_skip = (const float*)d_in[11];
    const float* W_out  = (const float*)d_in[12];
    const float* W1     = (const float*)d_in[13];
    const float* b1     = (const float*)d_in[14];
    const float* W2     = (const float*)d_in[15];
    const float* b2     = (const float*)d_in[16];
    float* out = (float*)d_out;

    const size_t M = M_ROWS;
    // bf16 region
    __bf16* x     = (__bf16*)d_ws;                  // M*128
    __bf16* xln   = x     + M * 128;                // M*128
    __bf16* u     = xln   + M * 128;                // M*256 (scan -> y in place)
    __bf16* dbc   = u     + M * 256;                // M*40
    __bf16* hml   = dbc   + M * 40;                 // M*512
    __bf16* WinT  = hml   + M * 512;                // 4*512*128
    __bf16* WxpT  = WinT  + 4 * 512 * 128;          // 4*40*256
    __bf16* WdtT  = WxpT  + 4 * 40 * 256;           // 4*256*32 (K zero-padded)
    __bf16* WoutT = WdtT  + 4 * 256 * 32;           // 4*128*256
    __bf16* W1t   = WoutT + 4 * 128 * 256;          // 512*128
    __bf16* W2t   = W1t   + 512 * 128;              // 4096*512
    // fp32 region
    float* xz  = (float*)(W2t + (size_t)4096 * 512); // M*512
    float* dtb = xz + M * 512;                       // M*256
    float* wse = dtb + M * 256;
    if (ws_size < (size_t)((char*)wse - (char*)d_ws)) return;

    const dim3 gB(32, 4);
    auto blocks = [](int Mm, int Nn) {
        int t = ((Mm + 31) / 32) * ((Nn + 63) / 64);
        return (t + 3) / 4;
    };
    auto cb256 = [](int n) { return (n + 255) / 256; };

    // weight bf16 + transpose (+K pad) pre-pass
    for (int l = 0; l < N_LAYERS; ++l) {
        k_cvtT<<<cb256(512 * 128), 256, 0, stream>>>(
            W_in + (size_t)l * 128 * 512, WinT + (size_t)l * 512 * 128, 128, 512, 128);
        k_cvtT<<<cb256(40 * 256), 256, 0, stream>>>(
            W_xp + (size_t)l * 256 * 40, WxpT + (size_t)l * 40 * 256, 256, 40, 256);
        k_cvtT<<<cb256(256 * 32), 256, 0, stream>>>(
            W_dt + (size_t)l * 8 * 256, WdtT + (size_t)l * 256 * 32, 8, 256, 32);
        k_cvtT<<<cb256(128 * 256), 256, 0, stream>>>(
            W_out + (size_t)l * 256 * 128, WoutT + (size_t)l * 128 * 256, 256, 128, 256);
    }
    k_cvtT<<<cb256(512 * 128), 256, 0, stream>>>(W1, W1t, 128, 512, 128);
    k_cvtT<<<cb256(4096 * 512), 256, 0, stream>>>(W2, W2t, 512, 4096, 512);

    k_embed<<<cb256((int)(M * 128)), 256, 0, stream>>>(tokens, emb, x, (int)(M * 128));

    for (int l = 0; l < N_LAYERS; ++l) {
        k_layernorm<<<(int)((M + 7) / 8), dim3(32, 8), 0, stream>>>(
            x, xln, ln_w + l * D_MODEL, ln_b + l * D_MODEL, (int)M);
        // xz = ln(x) @ W_in  (128 -> 512), fp32 out (conv/gate read it)
        gemm_bf16_wmma<0, float><<<blocks(M, 512), gB, 0, stream>>>(
            xln, WinT + (size_t)l * 512 * 128, xz, nullptr, (int)M, 512, 128, 128, 512);
        // u = silu(conv1d(xc))  -> bf16
        k_conv_silu<<<cb256((int)(M * 256)), 256, 0, stream>>>(
            xz, conv_w + (size_t)l * D_INNER * 4, conv_b + (size_t)l * D_INNER,
            u, (int)(M * 256));
        // dbc = u @ W_xproj  (256 -> 40) -> bf16
        gemm_bf16_wmma<0, __bf16><<<blocks(M, 40), gB, 0, stream>>>(
            u, WxpT + (size_t)l * 40 * 256, dbc, nullptr, (int)M, 40, 256, 256, 40);
        // dt = softplus(dbc[:, :8] @ W_dt + b_dt); K padded to 32 w/ zero rows
        gemm_bf16_wmma<3, float><<<blocks(M, 256), gB, 0, stream>>>(
            dbc, WdtT + (size_t)l * 256 * 32, dtb, b_dt + (size_t)l * D_INNER,
            (int)M, 256, 32, 40, 256);
        // sequential selective scan (+ D skip + silu(z) gate), y in place of u
        k_scan<<<BB, 256, 0, stream>>>(
            dtb, dbc, xz, A_log + (size_t)l * D_INNER * D_STATE,
            D_skip + (size_t)l * D_INNER, u);
        // x = y @ W_out  (256 -> 128) -> bf16
        gemm_bf16_wmma<0, __bf16><<<blocks(M, 128), gB, 0, stream>>>(
            u, WoutT + (size_t)l * 128 * 256, x, nullptr, (int)M, 128, 256, 256, 128);
    }

    // MLP head: relu(x @ W1 + b1) @ W2 + b2
    gemm_bf16_wmma<2, __bf16><<<blocks(M, D_FF), gB, 0, stream>>>(
        x, W1t, hml, b1, (int)M, D_FF, 128, 128, D_FF);
    gemm_bf16_wmma<1, float><<<blocks(M, 4096), gB, 0, stream>>>(
        hml, W2t, out, b2, (int)M, 4096, D_FF, D_FF, 4096);
}
